// GraphTransformer_88338887344527
// MI455X (gfx1250) — matrix operand order repaired and checked by reading, not compiled, not run on hardware
//
#include <hip/hip_runtime.h>
#include <hip/hip_bf16.h>

typedef __attribute__((ext_vector_type(16))) __bf16 v16bf;
typedef __attribute__((ext_vector_type(8)))  __bf16 v8bf;
typedef __attribute__((ext_vector_type(8)))  float  v8f;

#define TPB 256
#define GK  128           // K is 128 for every GEMM in this model
#define LDSPAD 8          // +16B per row: conflict-free-ish ds_load_b128
#define LDSTRIDE (GK + LDSPAD)   // 136 bf16 = 272B per staged row

// ---------------------------------------------------------------------------
// Elementwise helpers
// ---------------------------------------------------------------------------
__global__ __launch_bounds__(TPB) void cast_f32_to_bf16(const float* __restrict__ in,
                                                        __bf16* __restrict__ out,
                                                        long long n) {
    long long i = (long long)blockIdx.x * blockDim.x + threadIdx.x;
    long long stride = (long long)gridDim.x * blockDim.x;
    for (; i < n; i += stride) out[i] = (__bf16)in[i];
}

__global__ __launch_bounds__(TPB) void fill_f32(float* __restrict__ p, float v, long long n) {
    long long i = (long long)blockIdx.x * blockDim.x + threadIdx.x;
    long long stride = (long long)gridDim.x * blockDim.x;
    for (; i < n; i += stride) p[i] = v;
}

// ReLU in place + cast to bf16 for the next layer's WMMA GEMMs
__global__ __launch_bounds__(TPB) void relu_and_cast(float* __restrict__ h,
                                                     __bf16* __restrict__ hb,
                                                     long long n) {
    long long i = (long long)blockIdx.x * blockDim.x + threadIdx.x;
    long long stride = (long long)gridDim.x * blockDim.x;
    for (; i < n; i += stride) {
        float v = h[i];
        v = v > 0.0f ? v : 0.0f;
        h[i] = v;
        hb[i] = (__bf16)v;
    }
}

// ---------------------------------------------------------------------------
// Pack W[K,N] (f32, row-major) into WMMA B-fragment order (bf16):
//   Wp[((tileN*(K/32) + ks)*32 + lane)*16 + i] = W[k, col]
//   col = tileN*16 + (lane&15); k = ks*32 + (lane>>4)*8 + (i&7) + (i>>3)*16
// After packing, each lane's 16-element B fragment is 32 contiguous bytes.
// ---------------------------------------------------------------------------
__global__ __launch_bounds__(TPB) void pack_w_bf16(const float* __restrict__ W,
                                                   __bf16* __restrict__ Wp,
                                                   int K, int N) {
    const int ksteps = K >> 5;
    long long t = (long long)blockIdx.x * blockDim.x + threadIdx.x;
    long long stride = (long long)gridDim.x * blockDim.x;
    const long long total = (long long)K * N;
    for (; t < total; t += stride) {
        const int i    = (int)(t & 15);
        const int lane = (int)((t >> 4) & 31);
        const long long grp = t >> 9;               // tileN*ksteps + ks
        const int ks    = (int)(grp % ksteps);
        const int tileN = (int)(grp / ksteps);
        const int col = tileN * 16 + (lane & 15);
        const int k   = ks * 32 + (lane >> 4) * 8 + (i & 7) + (i >> 3) * 16;
        Wp[t] = (__bf16)W[(long long)k * N + col];
    }
}

// ---------------------------------------------------------------------------
// bf16 WMMA GEMM: C[M,N] = A[M,128] @ W[128,N] + bias[N]  (fp32 accumulate)
// Block = 8 waves. The block's A panel (rowsPerBlock*16 rows x 128 K) is
// staged once in LDS (coalesced b128 loads, +16B row pad), then each wave
// computes TWO adjacent 16x16 N-tiles sharing one A fragment per k-step.
//   N=128: 4 tile-pairs/row, 2 tileM rows/block (32 A rows)
//   N= 64: 2 tile-pairs/row, 4 tileM rows/block (64 A rows)
// ---------------------------------------------------------------------------
__global__ __launch_bounds__(TPB) void gemm_bf16_wmma(const __bf16* __restrict__ A,
                                                      const __bf16* __restrict__ Wp,
                                                      const float* __restrict__ bias,
                                                      float* __restrict__ C,
                                                      int M, int N) {
    __shared__ __align__(16) __bf16 sA[64 * LDSTRIDE];   // up to 64 rows staged

    const int tilesN       = N >> 4;
    const int pairsPerRow  = tilesN >> 1;                // waves per tileM row
    const int rowsPerBlock = 8 / pairsPerRow;            // tileM rows per block
    const int R            = rowsPerBlock * 16;          // A rows staged

    const int tid  = threadIdx.x;
    const long long gRow0 = (long long)blockIdx.x * R;

    // ---- cooperative A panel stage: R rows x 128 K bf16 (16B chunks) ----
    for (int j = 0; j < rowsPerBlock; ++j) {
        const int ci = j * TPB + tid;                    // 16 chunks per row
        const int r  = ci >> 4;
        const int c8 = (ci & 15) * 8;
        const long long gr = gRow0 + r;
        v8bf vreg;
        if (gr < M) {
            vreg = *(const v8bf*)(A + gr * GK + c8);
        } else {
            #pragma unroll
            for (int i = 0; i < 8; ++i) vreg[i] = (__bf16)0.0f;
        }
        *(v8bf*)(&sA[r * LDSTRIDE + c8]) = vreg;
    }
    __syncthreads();

    // ---- per-wave compute: 2 N-tiles, shared A fragment ----
    const int lane   = tid & 31;
    const int waveId = tid >> 5;
    const int wr     = waveId / pairsPerRow;             // tileM row within block
    const int wp     = waveId % pairsPerRow;             // tileN pair
    const int tileM  = blockIdx.x * rowsPerBlock + wr;
    if (tileM * 16 >= M) return;                         // wave-uniform (after barrier)

    const int r16  = lane & 15;
    const int half = lane >> 4;
    const int tileN0 = wp * 2;

    const __bf16* as  = &sA[(wr * 16 + r16) * LDSTRIDE + half * 8];
    const __bf16* bb0 = Wp + ((long long)tileN0 * (GK >> 5) * 32 + lane) * 16;
    const __bf16* bb1 = bb0 + (long long)(GK >> 5) * 32 * 16;

    v8f acc0 = {}, acc1 = {};
    #pragma unroll
    for (int ks = 0; ks < (GK >> 5); ++ks) {             // 4 steps of 32
        // A fragment 16x32 from LDS (ds_load_b128 x2)
        const v8bf a0 = *(const v8bf*)(as + ks * 32);
        const v8bf a1 = *(const v8bf*)(as + ks * 32 + 16);
        v16bf a;
        #pragma unroll
        for (int i = 0; i < 8; ++i) { a[i] = a0[i]; a[8 + i] = a1[i]; }

        // Two pre-packed B fragments: 32 contiguous bytes per lane each
        const v16bf b0 = *(const v16bf*)(bb0 + ks * 512);
        const v16bf b1 = *(const v16bf*)(bb1 + ks * 512);

        acc0 = __builtin_amdgcn_wmma_f32_16x16x32_bf16(false, a, false, b0, (short)0, acc0, false, false);
        acc1 = __builtin_amdgcn_wmma_f32_16x16x32_bf16(false, a, false, b1, (short)0, acc1, false, false);
    }

    // C fragment: VGPR r, lane -> (M = r + 8*half, N = lane&15)
    const int col0 = tileN0 * 16 + r16;
    const float bv0 = bias[col0];
    const float bv1 = bias[col0 + 16];
    float* cp = C + (long long)(tileM * 16 + half * 8) * N + col0;
    #pragma unroll
    for (int r = 0; r < 8; ++r) {
        cp[(long long)r * N]      = acc0[r] + bv0;
        cp[(long long)r * N + 16] = acc1[r] + bv1;
    }
}

// ---------------------------------------------------------------------------
// Edge logits fused with segment-max:
//   alpha[e,h] = <q[dst,h,:], k[src,h,:]> * scale ;  m[dst,h] = max(...)
// ---------------------------------------------------------------------------
__global__ __launch_bounds__(TPB) void attn_logits_max(const float* __restrict__ q,
                                                       const float* __restrict__ k,
                                                       const int* __restrict__ src,
                                                       const int* __restrict__ dst,
                                                       float* __restrict__ alpha,
                                                       float* __restrict__ m,
                                                       long long E, int H, int Cc, float scale) {
    long long t = (long long)blockIdx.x * blockDim.x + threadIdx.x;
    long long stride = (long long)gridDim.x * blockDim.x;
    const long long total = E * H;
    const int D = H * Cc;
    for (; t < total; t += stride) {
        const long long e = t / H;
        const int h = (int)(t - e * H);
        const int s = src[e], d = dst[e];
        const float4* qp = (const float4*)(q + (long long)d * D + h * Cc);
        const float4* kp = (const float4*)(k + (long long)s * D + h * Cc);
        float acc = 0.0f;
        #pragma unroll 4
        for (int c = 0; c < (Cc >> 2); ++c) {
            const float4 qv = qp[c];
            const float4 kv = kp[c];
            acc += qv.x * kv.x + qv.y * kv.y + qv.z * kv.z + qv.w * kv.w;
        }
        acc *= scale;
        alpha[t] = acc;
        unsafeAtomicMax(&m[(long long)d * H + h], acc);   // global_atomic_max_num_f32
    }
}

// ex[e,h] = exp(alpha - m[dst]); denom[dst,h] += ex   (in-place over alpha)
__global__ __launch_bounds__(TPB) void exp_and_denom(float* __restrict__ alpha,
                                                     const float* __restrict__ m,
                                                     const int* __restrict__ dst,
                                                     float* __restrict__ denom,
                                                     long long E, int H) {
    long long t = (long long)blockIdx.x * blockDim.x + threadIdx.x;
    long long stride = (long long)gridDim.x * blockDim.x;
    const long long total = E * H;
    for (; t < total; t += stride) {
        const long long e = t / H;
        const int h = (int)(t - e * H);
        const long long di = (long long)dst[e] * H + h;
        const float ex = __expf(alpha[t] - m[di]);
        alpha[t] = ex;
        unsafeAtomicAdd(&denom[di], ex);                  // global_atomic_add_f32
    }
}

// out[dst, c..c+3] += v[src, c..c+3] * ex[e,h] / denom[dst,h]
// One thread per (edge, 4 channels); out pre-initialized with the skip term.
__global__ __launch_bounds__(TPB) void aggregate(const float* __restrict__ v,
                                                 const float* __restrict__ ex,
                                                 const float* __restrict__ denom,
                                                 const int* __restrict__ src,
                                                 const int* __restrict__ dst,
                                                 float* __restrict__ out,
                                                 long long E, int H, int Cc) {
    const int D = H * Cc;
    const int D4 = D >> 2;
    long long t = (long long)blockIdx.x * blockDim.x + threadIdx.x;
    long long stride = (long long)gridDim.x * blockDim.x;
    const long long total = E * D4;
    for (; t < total; t += stride) {
        const long long e = t / D4;
        const int c4 = (int)(t - e * D4);
        const int c  = c4 << 2;
        const int h  = c / Cc;
        const int s = src[e], d = dst[e];
        const float a = ex[e * H + h] / denom[(long long)d * H + h];
        const float4 vv = *(const float4*)(v + (long long)s * D + c);
        float* op = out + (long long)d * D + c;
        unsafeAtomicAdd(op + 0, vv.x * a);
        unsafeAtomicAdd(op + 1, vv.y * a);
        unsafeAtomicAdd(op + 2, vv.z * a);
        unsafeAtomicAdd(op + 3, vv.w * a);
    }
}

// ---------------------------------------------------------------------------
// Host-side orchestration
// ---------------------------------------------------------------------------
static inline int gsblocks(long long n) {
    long long b = (n + TPB - 1) / TPB;
    if (b > 262144) b = 262144;
    if (b < 1) b = 1;
    return (int)b;
}

extern "C" void kernel_launch(void* const* d_in, const int* in_sizes, int n_in,
                              void* d_out, int out_size, void* d_ws, size_t ws_size,
                              hipStream_t stream) {
    const float* x   = (const float*)d_in[0];
    const int*   ei  = (const int*)d_in[1];
    const float *Wq1 = (const float*)d_in[2],  *bq1 = (const float*)d_in[3];
    const float *Wk1 = (const float*)d_in[4],  *bk1 = (const float*)d_in[5];
    const float *Wv1 = (const float*)d_in[6],  *bv1 = (const float*)d_in[7];
    const float *Ws1 = (const float*)d_in[8],  *bs1 = (const float*)d_in[9];
    const float *Wq2 = (const float*)d_in[10], *bq2 = (const float*)d_in[11];
    const float *Wk2 = (const float*)d_in[12], *bk2 = (const float*)d_in[13];
    const float *Wv2 = (const float*)d_in[14], *bv2 = (const float*)d_in[15];
    const float *Ws2 = (const float*)d_in[16], *bs2 = (const float*)d_in[17];

    const int IN = 128, HID = 128, OUT = 64, H1 = 8, C1 = 16, C2 = 64;
    const int N = in_sizes[0] / IN;           // 100000
    const long long E = in_sizes[1] / 2;      // 1600000
    const int* src = ei;
    const int* dst = ei + E;

    // ---- workspace carve-out ----
    char* ws = (char*)d_ws;
    size_t off = 0;
    auto carve = [&](size_t bytes) -> void* {
        void* p = ws + off;
        off = (off + bytes + 255) & ~(size_t)255;
        return p;
    };
    __bf16* xb   = (__bf16*)carve((size_t)N * IN * sizeof(__bf16));
    __bf16* wq1p = (__bf16*)carve((size_t)IN * HID * sizeof(__bf16));
    __bf16* wk1p = (__bf16*)carve((size_t)IN * HID * sizeof(__bf16));
    __bf16* wv1p = (__bf16*)carve((size_t)IN * HID * sizeof(__bf16));
    __bf16* ws1p = (__bf16*)carve((size_t)IN * HID * sizeof(__bf16));
    __bf16* wq2p = (__bf16*)carve((size_t)HID * OUT * sizeof(__bf16));
    __bf16* wk2p = (__bf16*)carve((size_t)HID * OUT * sizeof(__bf16));
    __bf16* wv2p = (__bf16*)carve((size_t)HID * OUT * sizeof(__bf16));
    __bf16* ws2p = (__bf16*)carve((size_t)HID * OUT * sizeof(__bf16));
    float* q1    = (float*)carve((size_t)N * HID * sizeof(float));
    float* k1    = (float*)carve((size_t)N * HID * sizeof(float));
    float* v1    = (float*)carve((size_t)N * HID * sizeof(float));
    float* h     = (float*)carve((size_t)N * HID * sizeof(float));   // skip + aggregation
    __bf16* hb   = (__bf16*)carve((size_t)N * HID * sizeof(__bf16));
    float* al1   = (float*)carve((size_t)E * H1 * sizeof(float));    // logits -> exp in place
    float* m1    = (float*)carve((size_t)N * H1 * sizeof(float));
    float* den1  = (float*)carve((size_t)N * H1 * sizeof(float));
    float* q2    = (float*)carve((size_t)N * OUT * sizeof(float));
    float* k2    = (float*)carve((size_t)N * OUT * sizeof(float));
    float* v2    = (float*)carve((size_t)N * OUT * sizeof(float));
    float* al2   = (float*)carve((size_t)E * sizeof(float));
    float* m2    = (float*)carve((size_t)N * sizeof(float));
    float* den2  = (float*)carve((size_t)N * sizeof(float));
    float* out   = (float*)d_out;             // [N, OUT]

    // ---- 0) cast x; pack all weights into WMMA B-fragment order (bf16) ----
    long long nx = (long long)N * IN;
    cast_f32_to_bf16<<<gsblocks(nx), TPB, 0, stream>>>(x, xb, nx);
    pack_w_bf16<<<gsblocks(IN * HID), TPB, 0, stream>>>(Wq1, wq1p, IN, HID);
    pack_w_bf16<<<gsblocks(IN * HID), TPB, 0, stream>>>(Wk1, wk1p, IN, HID);
    pack_w_bf16<<<gsblocks(IN * HID), TPB, 0, stream>>>(Wv1, wv1p, IN, HID);
    pack_w_bf16<<<gsblocks(IN * HID), TPB, 0, stream>>>(Ws1, ws1p, IN, HID);
    pack_w_bf16<<<gsblocks(HID * OUT), TPB, 0, stream>>>(Wq2, wq2p, HID, OUT);
    pack_w_bf16<<<gsblocks(HID * OUT), TPB, 0, stream>>>(Wk2, wk2p, HID, OUT);
    pack_w_bf16<<<gsblocks(HID * OUT), TPB, 0, stream>>>(Wv2, wv2p, HID, OUT);
    pack_w_bf16<<<gsblocks(HID * OUT), TPB, 0, stream>>>(Ws2, ws2p, HID, OUT);

    const int tilesM = N / 16;                // 6250

    // ---- layer 1 GEMMs (WMMA): q/k/v + skip (skip lands in h) ----
    {
        const int rowsPerBlock = 8 / ((HID / 16) / 2);     // 2
        const int blocks = (tilesM + rowsPerBlock - 1) / rowsPerBlock;
        gemm_bf16_wmma<<<blocks, TPB, 0, stream>>>(xb, wq1p, bq1, q1, N, HID);
        gemm_bf16_wmma<<<blocks, TPB, 0, stream>>>(xb, wk1p, bk1, k1, N, HID);
        gemm_bf16_wmma<<<blocks, TPB, 0, stream>>>(xb, wv1p, bv1, v1, N, HID);
        gemm_bf16_wmma<<<blocks, TPB, 0, stream>>>(xb, ws1p, bs1, h,  N, HID);
    }

    // ---- layer 1 edge softmax + aggregation ----
    fill_f32<<<gsblocks((long long)N * H1), TPB, 0, stream>>>(m1, -__builtin_huge_valf(), (long long)N * H1);
    fill_f32<<<gsblocks((long long)N * H1), TPB, 0, stream>>>(den1, 0.0f, (long long)N * H1);
    attn_logits_max<<<gsblocks(E * H1), TPB, 0, stream>>>(q1, k1, src, dst, al1, m1, E, H1, C1, 0.25f);
    exp_and_denom<<<gsblocks(E * H1), TPB, 0, stream>>>(al1, m1, dst, den1, E, H1);
    aggregate<<<gsblocks(E * (HID / 4)), TPB, 0, stream>>>(v1, al1, den1, src, dst, h, E, H1, C1);

    // ---- ReLU + bf16 cast ----
    relu_and_cast<<<gsblocks(nx), TPB, 0, stream>>>(h, hb, nx);

    // ---- layer 2 GEMMs (WMMA): q/k/v + skip (skip lands in d_out) ----
    {
        const int rowsPerBlock = 8 / ((OUT / 16) / 2);     // 4
        const int blocks = (tilesM + rowsPerBlock - 1) / rowsPerBlock;
        gemm_bf16_wmma<<<blocks, TPB, 0, stream>>>(hb, wq2p, bq2, q2, N, OUT);
        gemm_bf16_wmma<<<blocks, TPB, 0, stream>>>(hb, wk2p, bk2, k2, N, OUT);
        gemm_bf16_wmma<<<blocks, TPB, 0, stream>>>(hb, wv2p, bv2, v2, N, OUT);
        gemm_bf16_wmma<<<blocks, TPB, 0, stream>>>(hb, ws2p, bs2, out, N, OUT);
    }

    // ---- layer 2 edge softmax + aggregation (H=1, mean over 1 head == identity) ----
    fill_f32<<<gsblocks(N), TPB, 0, stream>>>(m2, -__builtin_huge_valf(), N);
    fill_f32<<<gsblocks(N), TPB, 0, stream>>>(den2, 0.0f, N);
    attn_logits_max<<<gsblocks(E), TPB, 0, stream>>>(q2, k2, src, dst, al2, m2, E, 1, C2, 0.125f);
    exp_and_denom<<<gsblocks(E), TPB, 0, stream>>>(al2, m2, dst, den2, E, 1);
    aggregate<<<gsblocks(E * (OUT / 4)), TPB, 0, stream>>>(v2, al2, den2, src, dst, out, E, 1, C2);
}
